// HermesBlock_67568425501242
// MI455X (gfx1250) — compile-verified, matrix operand order checked by reading, not run on hardware
//
#include <hip/hip_runtime.h>

typedef __attribute__((ext_vector_type(16))) _Float16 v16h;
typedef __attribute__((ext_vector_type(8)))  float    v8f;

union H16 { v16h v; _Float16 h[16]; };
union F8  { v8f  v; float    f[8];  };

constexpr int kB = 4, kT = 1024, kD = 1024, kH = 16, kS = 64, kHid = 3072;
constexpr int kM = kB * kT;              // 4096 token rows

// ---------------------------------------------------------------------------
// WMMA + async-copy helpers (gfx1250)
// ---------------------------------------------------------------------------
__device__ inline v8f wmma32(v16h a, v16h b, v8f c) {
  return __builtin_amdgcn_wmma_f32_16x16x32_f16(false, a, false, b, (short)0, c,
                                                false, false);
}

__device__ inline v8f zero8() {
  v8f z = {0.f, 0.f, 0.f, 0.f, 0.f, 0.f, 0.f, 0.f};
  return z;
}

// Low 32 bits of a generic pointer to LDS == LDS byte address (ISA §10.2).
__device__ inline unsigned lds_addr(const void* p) {
  return (unsigned)(unsigned long long)p;
}

// One 16B async copy global -> LDS per active lane (ASYNCcnt-tracked).
__device__ inline void async_copy16(unsigned lds, const void* g) {
  asm volatile("global_load_async_to_lds_b128 %0, %1, off"
               :: "v"(lds), "v"((unsigned long long)g)
               : "memory");
}

__device__ inline void wait_async0() {
  asm volatile("s_wait_asynccnt 0x0" ::: "memory");
}
__device__ inline void wait_async4() {
  asm volatile("s_wait_asynccnt 0x4" ::: "memory");
}

// A-fragment (16x32, row-major source with leading dim `ld`), per ISA layout:
// lanes 0-15 hold K 0..7 & 16..23, lanes 16-31 hold K 8..15 & 24..31.
// Per lane this is two contiguous 16B chunks -> compiler emits ds_load_b128 x2.
__device__ inline v16h frag_row(const _Float16* base, int ld, int lane) {
  H16 u;
  int row = lane & 15;
  int kb  = (lane & 16) >> 1;      // 0 or 8
  const _Float16* p = base + row * ld;
#pragma unroll
  for (int i = 0; i < 8; ++i) {
    int k = ((i & 4) << 2) + kb + ((i & 3) << 1);
    u.h[2 * i]     = p[k];
    u.h[2 * i + 1] = p[k + 1];
  }
  return u.v;
}

// B-fragment where source is K-major (rows = K, cols = N): lane n reads col n.
__device__ inline v16h frag_col(const _Float16* base, int ld, int lane) {
  H16 u;
  int col = lane & 15;
  int kb  = (lane & 16) >> 1;
#pragma unroll
  for (int i = 0; i < 8; ++i) {
    int k = ((i & 4) << 2) + kb + ((i & 3) << 1);
    u.h[2 * i]     = base[k * ld + col];
    u.h[2 * i + 1] = base[(k + 1) * ld + col];
  }
  return u.v;
}

// ---------------------------------------------------------------------------
// Weight quantization: per-tensor symmetric 6-bit fake-quant -> f16
// ---------------------------------------------------------------------------
__global__ void init_scales_kernel(unsigned* p) {
  if (threadIdx.x < 8) p[threadIdx.x] = 0u;
}

__global__ __launch_bounds__(256) void absmax_kernel(
    const float* __restrict__ w, int n, unsigned* __restrict__ slot) {
  float m = 0.f;
  for (int i = blockIdx.x * 256 + threadIdx.x; i < n; i += gridDim.x * 256)
    m = fmaxf(m, fabsf(w[i]));
  __shared__ float s[256];
  s[threadIdx.x] = m;
  __syncthreads();
  for (int st = 128; st > 0; st >>= 1) {
    if (threadIdx.x < st) s[threadIdx.x] = fmaxf(s[threadIdx.x], s[threadIdx.x + st]);
    __syncthreads();
  }
  if (threadIdx.x == 0) atomicMax(slot, __float_as_uint(s[0]));  // vals >= 0
}

__global__ __launch_bounds__(256) void quant_kernel(
    const float* __restrict__ w, int n, const unsigned* __restrict__ slot,
    _Float16* __restrict__ out) {
  float scale = __uint_as_float(*slot) * (1.0f / 31.0f) + 1e-8f;
  for (int i = blockIdx.x * 256 + threadIdx.x; i < n; i += gridDim.x * 256) {
    float q = rintf(w[i] / scale);           // round-half-even, matches jnp.round
    q = fminf(fmaxf(q, -31.f), 31.f);
    out[i] = (_Float16)(q * scale);
  }
}

// eff[h*64+i, d] = sum_j q6(su)[i,j] * q6(w)[h*64+j, d]   (folds stalk map)
__global__ __launch_bounds__(256) void compose_kernel(
    const _Float16* __restrict__ sm, const _Float16* __restrict__ wm,
    _Float16* __restrict__ eff) {
  int i   = blockIdx.x * 256 + threadIdx.x;   // over D*D
  int row = i >> 10;
  int col = i & 1023;
  int h  = row >> 6;
  int si = row & 63;
  const _Float16* sp = sm + si * 64;
  const _Float16* wp = wm + (size_t)(h * 64) * kD + col;
  float acc = 0.f;
#pragma unroll 8
  for (int j = 0; j < 64; ++j) acc += (float)sp[j] * (float)wp[(size_t)j * kD];
  eff[i] = (_Float16)acc;
}

// ---------------------------------------------------------------------------
// LayerNorm (f32 in, f16 out), one block per row
// ---------------------------------------------------------------------------
__global__ __launch_bounds__(256) void ln_kernel(
    const float* __restrict__ x, const float* __restrict__ g,
    const float* __restrict__ bb, _Float16* __restrict__ outh, int Dn) {
  int row = blockIdx.x;
  const float* xr = x + (size_t)row * Dn;
  int tid = threadIdx.x;
  float sum = 0.f, sq = 0.f;
  for (int i = tid; i < Dn; i += 256) {
    float v = xr[i];
    sum += v;
    sq  += v * v;
  }
  __shared__ float s1[256], s2[256];
  s1[tid] = sum; s2[tid] = sq;
  __syncthreads();
  for (int st = 128; st > 0; st >>= 1) {
    if (tid < st) { s1[tid] += s1[tid + st]; s2[tid] += s2[tid + st]; }
    __syncthreads();
  }
  float mu   = s1[0] / Dn;
  float var  = s2[0] / Dn - mu * mu;
  float rstd = rsqrtf(var + 1e-5f);
  for (int i = tid; i < Dn; i += 256)
    outh[(size_t)row * Dn + i] = (_Float16)((xr[i] - mu) * rstd * g[i] + bb[i]);
}

// SO(2) rotate pairs -> LN over HID -> SiLU -> f16, one block per row
__global__ __launch_bounds__(256) void rot_ln_silu_kernel(
    const float* __restrict__ hbuf, const float* __restrict__ theta,
    const float* __restrict__ g, const float* __restrict__ bb,
    _Float16* __restrict__ outh) {
  int row = blockIdx.x;
  const float* hr = hbuf + (size_t)row * kHid;
  int tid = threadIdx.x;
  float rv[12];
  float sum = 0.f, sq = 0.f;
#pragma unroll
  for (int i = 0; i < 6; ++i) {
    int p = tid + i * 256;                    // pair index, < 1536
    float a = hr[2 * p], b = hr[2 * p + 1];
    float c = cosf(theta[p]), s = sinf(theta[p]);
    float ra = c * a - s * b;
    float rb = s * a + c * b;
    rv[2 * i] = ra; rv[2 * i + 1] = rb;
    sum += ra + rb;
    sq  += ra * ra + rb * rb;
  }
  __shared__ float s1[256], s2[256];
  s1[tid] = sum; s2[tid] = sq;
  __syncthreads();
  for (int st = 128; st > 0; st >>= 1) {
    if (tid < st) { s1[tid] += s1[tid + st]; s2[tid] += s2[tid + st]; }
    __syncthreads();
  }
  float mu   = s1[0] / kHid;
  float var  = s2[0] / kHid - mu * mu;
  float rstd = rsqrtf(var + 1e-5f);
#pragma unroll
  for (int i = 0; i < 6; ++i) {
    int p = tid + i * 256;
#pragma unroll
    for (int j = 0; j < 2; ++j) {
      int idx = 2 * p + j;
      float v = (rv[2 * i + j] - mu) * rstd * g[idx] + bb[idx];
      float act = v / (1.f + __expf(-v));     // silu
      outh[(size_t)row * kHid + idx] = (_Float16)act;
    }
  }
}

// ---------------------------------------------------------------------------
// WMMA GEMM with async double-buffered LDS staging:
//   C[M,N] = A[M,K] @ W[N,K]^T (+ optional f32 residual)
// Block: 256 threads = 8 waves; C tile 128x128; wave tile 32x64; K-step 32.
// Each thread issues 4 async 16B copies per K-step (2 A chunks + 2 W chunks).
// ---------------------------------------------------------------------------
#define LDT 40   // LDS row pitch in halves (80B, 16B aligned, decent banking)
__global__ __launch_bounds__(256) void gemm_kernel(
    const _Float16* __restrict__ A, const _Float16* __restrict__ W,
    const float* __restrict__ resid, float* __restrict__ Cf,
    _Float16* __restrict__ Ch, int N, int K) {
  __shared__ _Float16 As[2][128 * LDT];
  __shared__ _Float16 Ws[2][128 * LDT];
  int tid  = threadIdx.x;
  int wave = tid >> 5;
  int lane = tid & 31;
  int mBase = blockIdx.y * 128;
  int nBase = blockIdx.x * 128;
  int wrow  = (wave >> 1) * 32;   // wave row offset within tile
  int wcol  = (wave & 1) * 64;    // wave col offset within tile

  // Per-thread copy chunks: tile is 128 rows x 32 halves = 512 x 16B chunks.
  int c0 = tid, c1 = tid + 256;
  int r0 = c0 >> 2, o0 = (c0 & 3) << 3;
  int r1 = c1 >> 2, o1 = (c1 & 3) << 3;
  const _Float16* aG0 = A + (size_t)(mBase + r0) * K + o0;
  const _Float16* aG1 = A + (size_t)(mBase + r1) * K + o1;
  const _Float16* wG0 = W + (size_t)(nBase + r0) * K + o0;
  const _Float16* wG1 = W + (size_t)(nBase + r1) * K + o1;
  unsigned aL0[2], aL1[2], wL0[2], wL1[2];
#pragma unroll
  for (int bf = 0; bf < 2; ++bf) {
    aL0[bf] = lds_addr(&As[bf][r0 * LDT + o0]);
    aL1[bf] = lds_addr(&As[bf][r1 * LDT + o1]);
    wL0[bf] = lds_addr(&Ws[bf][r0 * LDT + o0]);
    wL1[bf] = lds_addr(&Ws[bf][r1 * LDT + o1]);
  }

  v8f acc[2][4];
#pragma unroll
  for (int m = 0; m < 2; ++m)
#pragma unroll
    for (int j = 0; j < 4; ++j) acc[m][j] = zero8();

  // prologue: stage K-step 0 into buffer 0
  async_copy16(aL0[0], aG0);
  async_copy16(aL1[0], aG1);
  async_copy16(wL0[0], wG0);
  async_copy16(wL1[0], wG1);

  int nsteps = K >> 5;
  for (int s = 0; s < nsteps; ++s) {
    int cur = s & 1;
    if (s + 1 < nsteps) {
      int kk = (s + 1) << 5;
      async_copy16(aL0[cur ^ 1], aG0 + kk);
      async_copy16(aL1[cur ^ 1], aG1 + kk);
      async_copy16(wL0[cur ^ 1], wG0 + kk);
      async_copy16(wL1[cur ^ 1], wG1 + kk);
      wait_async4();   // oldest 4 (current tile) complete; next 4 in flight
    } else {
      wait_async0();
    }
    __syncthreads();

    const _Float16* ab = &As[cur][wrow * LDT];
    const _Float16* wb = &Ws[cur][wcol * LDT];
    v16h a0 = frag_row(ab, LDT, lane);
    v16h a1 = frag_row(ab + 16 * LDT, LDT, lane);
#pragma unroll
    for (int j = 0; j < 4; ++j) {
      v16h bf = frag_row(wb + j * 16 * LDT, LDT, lane);
      acc[0][j] = wmma32(a0, bf, acc[0][j]);
      acc[1][j] = wmma32(a1, bf, acc[1][j]);
    }
    __syncthreads();   // safe to overwrite `cur` buffer at step s+2
  }

  // C/D layout: VGPR r <-> M = r + 8*(lane>=16), N = lane&15
  int roff = (lane & 16) >> 1;
  int nc   = lane & 15;
#pragma unroll
  for (int m = 0; m < 2; ++m)
#pragma unroll
    for (int j = 0; j < 4; ++j) {
      F8 t; t.v = acc[m][j];
#pragma unroll
      for (int r = 0; r < 8; ++r) {
        size_t idx = (size_t)(mBase + wrow + m * 16 + r + roff) * N +
                     nBase + wcol + j * 16 + nc;
        float val = t.f[r];
        if (resid) val += resid[idx];
        if (Cf) Cf[idx] = val;
        if (Ch) Ch[idx] = (_Float16)val;
      }
    }
}

// ---------------------------------------------------------------------------
// Flash attention with 2-adic bias. Block = (b, h, 64 query rows), 4 waves.
// Qs/Ks/Vh are f16 [b*T + t, h*S + s] (ld = D). Output same layout.
// K/V tiles staged with async global->LDS copies.
// ---------------------------------------------------------------------------
#define ATT_LD 72
__global__ __launch_bounds__(128) void attn_kernel(
    const _Float16* __restrict__ Qs, const _Float16* __restrict__ Ks,
    const _Float16* __restrict__ Vh, const float* __restrict__ p_scale,
    _Float16* __restrict__ Ao) {
  __shared__ _Float16 Kt[64 * ATT_LD];
  __shared__ _Float16 Vt[64 * ATT_LD];
  __shared__ _Float16 Pt[4 * 16 * ATT_LD];

  int qb = blockIdx.x, h = blockIdx.y, b = blockIdx.z;
  int tid  = threadIdx.x;
  int wave = tid >> 5;
  int lane = tid & 31;
  int qbase = qb * 64;
  int roff  = (lane & 16) >> 1;   // +8 for upper half lanes
  int nc    = lane & 15;

  // per-thread async chunk assignment: 64x64 halves = 512 x 16B chunks, 4/thread
  int cr[4], co[4];
  unsigned kL[4], vL[4];
#pragma unroll
  for (int j = 0; j < 4; ++j) {
    int c = tid + j * 128;
    cr[j] = c >> 3;
    co[j] = (c & 7) << 3;
    kL[j] = lds_addr(&Kt[cr[j] * ATT_LD + co[j]]);
    vL[j] = lds_addr(&Vt[cr[j] * ATT_LD + co[j]]);
  }

  const _Float16* qptr =
      Qs + (size_t)(b * kT + qbase + wave * 16) * kD + h * kS;
  v16h a0 = frag_row(qptr, kD, lane);        // S = 0..31
  v16h a1 = frag_row(qptr + 32, kD, lane);   // S = 32..63

  float ps = p_scale[0];
  const float inv_sqrtS = 0.125f;            // 1/sqrt(64)

  v8f acc[4];
#pragma unroll
  for (int j = 0; j < 4; ++j) acc[j] = zero8();
  float m_run[8], l_run[8];
#pragma unroll
  for (int r = 0; r < 8; ++r) { m_run[r] = -1e30f; l_run[r] = 0.f; }

  for (int ub = 0; ub <= qb; ++ub) {
    int ubase = ub * 64;
    __syncthreads();   // previous iteration done reading Kt/Vt
    const _Float16* kg = Ks + (size_t)(b * kT + ubase) * kD + h * kS;
    const _Float16* vg = Vh + (size_t)(b * kT + ubase) * kD + h * kS;
#pragma unroll
    for (int j = 0; j < 4; ++j) {
      async_copy16(kL[j], kg + (size_t)cr[j] * kD + co[j]);
      async_copy16(vL[j], vg + (size_t)cr[j] * kD + co[j]);
    }
    wait_async0();
    __syncthreads();

    // scores: 4 x (16x16) tiles, contraction over S=64 in two WMMAs
    float sc[4][8];
#pragma unroll
    for (int j = 0; j < 4; ++j) {
      v16h b0 = frag_row(Kt + j * 16 * ATT_LD, ATT_LD, lane);
      v16h b1 = frag_row(Kt + j * 16 * ATT_LD + 32, ATT_LD, lane);
      v8f s = zero8();
      s = wmma32(a0, b0, s);
      s = wmma32(a1, b1, s);
      F8 t; t.v = s;
#pragma unroll
      for (int r = 0; r < 8; ++r) sc[j][r] = t.f[r];
    }

    // bias + causal mask + per-lane partial row max
    float pm[8];
#pragma unroll
    for (int r = 0; r < 8; ++r) pm[r] = -1e30f;
#pragma unroll
    for (int j = 0; j < 4; ++j) {
      int ucol = ubase + j * 16 + nc;
#pragma unroll
      for (int r = 0; r < 8; ++r) {
        int trow = qbase + wave * 16 + r + roff;
        float v = sc[j][r] * inv_sqrtS;
        int d = trow - ucol;
        float bias;
        if (d == 0) {
          bias = 1.0f;
        } else {
          int ad = d < 0 ? -d : d;
          int vp = __builtin_ctz(ad);
          vp = vp > 16 ? 16 : vp;
          bias = (float)vp * (1.0f / 16.0f);
        }
        v += ps * bias;
        if (ucol > trow) v = -1e30f;          // causal
        sc[j][r] = v;
        pm[r] = fmaxf(pm[r], v);
      }
    }
    // row max across the 16-lane half that owns each row
#pragma unroll
    for (int off = 1; off < 16; off <<= 1)
#pragma unroll
      for (int r = 0; r < 8; ++r)
        pm[r] = fmaxf(pm[r], __shfl_xor(pm[r], off, 32));

    float corr[8], rs[8];
#pragma unroll
    for (int r = 0; r < 8; ++r) {
      float nm = fmaxf(m_run[r], pm[r]);
      corr[r] = __expf(m_run[r] - nm);
      m_run[r] = nm;
      rs[r] = 0.f;
    }
#pragma unroll
    for (int j = 0; j < 4; ++j)
#pragma unroll
      for (int r = 0; r < 8; ++r) {
        float e = __expf(sc[j][r] - m_run[r]);
        sc[j][r] = e;
        rs[r] += e;
      }
#pragma unroll
    for (int off = 1; off < 16; off <<= 1)
#pragma unroll
      for (int r = 0; r < 8; ++r) rs[r] += __shfl_xor(rs[r], off, 32);
#pragma unroll
    for (int r = 0; r < 8; ++r) l_run[r] = l_run[r] * corr[r] + rs[r];
#pragma unroll
    for (int j = 0; j < 4; ++j) {
      F8 t; t.v = acc[j];
#pragma unroll
      for (int r = 0; r < 8; ++r) t.f[r] *= corr[r];
      acc[j] = t.v;
    }

    // re-stripe probabilities through LDS into A-fragment layout
    _Float16* pw = Pt + wave * 16 * ATT_LD;
#pragma unroll
    for (int j = 0; j < 4; ++j)
#pragma unroll
      for (int r = 0; r < 8; ++r)
        pw[(r + roff) * ATT_LD + j * 16 + nc] = (_Float16)sc[j][r];
    __syncthreads();

    v16h p0 = frag_row(pw, ATT_LD, lane);        // u = 0..31
    v16h p1 = frag_row(pw + 32, ATT_LD, lane);   // u = 32..63
#pragma unroll
    for (int j = 0; j < 4; ++j) {
      v16h vb0 = frag_col(Vt + j * 16, ATT_LD, lane);
      v16h vb1 = frag_col(Vt + 32 * ATT_LD + j * 16, ATT_LD, lane);
      acc[j] = wmma32(p0, vb0, acc[j]);
      acc[j] = wmma32(p1, vb1, acc[j]);
    }
  }

  // finalize: O / l, store as [b, t, h*S + s] f16
#pragma unroll
  for (int j = 0; j < 4; ++j) {
    F8 t; t.v = acc[j];
#pragma unroll
    for (int r = 0; r < 8; ++r) {
      size_t idx = (size_t)(b * kT + qbase + wave * 16 + r + roff) * kD +
                   h * kS + j * 16 + nc;
      Ao[idx] = (_Float16)(t.f[r] / l_run[r]);
    }
  }
}

// ---------------------------------------------------------------------------
// Host-side orchestration
// ---------------------------------------------------------------------------
extern "C" void kernel_launch(void* const* d_in, const int* in_sizes, int n_in,
                              void* d_out, int out_size, void* d_ws,
                              size_t ws_size, hipStream_t stream) {
  const float* x     = (const float*)d_in[0];
  const float* wq    = (const float*)d_in[1];
  const float* wk    = (const float*)d_in[2];
  const float* wv    = (const float*)d_in[3];
  const float* wo    = (const float*)d_in[4];
  const float* su    = (const float*)d_in[5];
  const float* sv    = (const float*)d_in[6];
  const float* pscal = (const float*)d_in[7];
  const float* n1g   = (const float*)d_in[8];
  const float* n1b   = (const float*)d_in[9];
  const float* n2g   = (const float*)d_in[10];
  const float* n2b   = (const float*)d_in[11];
  const float* wup   = (const float*)d_in[12];
  const float* wdn   = (const float*)d_in[13];
  const float* theta = (const float*)d_in[14];
  const float* mng   = (const float*)d_in[15];
  const float* mnb   = (const float*)d_in[16];
  float* out = (float*)d_out;

  char* ws = (char*)d_ws;
  size_t off = 0;
  auto carve = [&](size_t bytes) -> void* {
    void* p = ws + off;
    off = (off + bytes + 255) & ~(size_t)255;
    return p;
  };

  unsigned*  scales = (unsigned*)carve(64);                     // 8 slots
  _Float16*  q_wq   = (_Float16*)carve((size_t)kD * kD * 2);
  _Float16*  q_wk   = (_Float16*)carve((size_t)kD * kD * 2);
  _Float16*  q_wv   = (_Float16*)carve((size_t)kD * kD * 2);
  _Float16*  q_wo   = (_Float16*)carve((size_t)kD * kD * 2);
  _Float16*  q_su   = (_Float16*)carve((size_t)kS * kS * 2);
  _Float16*  q_sv   = (_Float16*)carve((size_t)kS * kS * 2);
  _Float16*  q_wup  = (_Float16*)carve((size_t)kHid * kD * 2);
  _Float16*  q_wdn  = (_Float16*)carve((size_t)kD * kHid * 2);
  _Float16*  eff_wq = (_Float16*)carve((size_t)kD * kD * 2);
  _Float16*  eff_wk = (_Float16*)carve((size_t)kD * kD * 2);
  _Float16*  xn1h   = (_Float16*)carve((size_t)kM * kD * 2);
  _Float16*  qs     = (_Float16*)carve((size_t)kM * kD * 2);
  _Float16*  ks     = (_Float16*)carve((size_t)kM * kD * 2);
  _Float16*  vh     = (_Float16*)carve((size_t)kM * kD * 2);
  _Float16*  ao     = (_Float16*)carve((size_t)kM * kD * 2);
  float*     x1     = (float*)carve((size_t)kM * kD * 4);
  _Float16*  xn2h   = (_Float16*)carve((size_t)kM * kD * 2);
  float*     hbuf   = (float*)carve((size_t)kM * kHid * 4);
  _Float16*  acth   = (_Float16*)carve((size_t)kM * kHid * 2);
  (void)ws_size; (void)in_sizes; (void)n_in; (void)out_size;

  // --- 1. quantize all weight tensors ---
  hipLaunchKernelGGL(init_scales_kernel, dim3(1), dim3(8), 0, stream, scales);
  const float* wts[8]  = {wq, wk, wv, wo, su, sv, wup, wdn};
  int          wn[8]   = {kD * kD, kD * kD, kD * kD, kD * kD,
                          kS * kS, kS * kS, kHid * kD, kD * kHid};
  _Float16*    wq16[8] = {q_wq, q_wk, q_wv, q_wo, q_su, q_sv, q_wup, q_wdn};
  for (int i = 0; i < 8; ++i) {
    int blocks = (wn[i] + 255) / 256;
    if (blocks > 512) blocks = 512;
    hipLaunchKernelGGL(absmax_kernel, dim3(blocks), dim3(256), 0, stream,
                       wts[i], wn[i], scales + i);
  }
  for (int i = 0; i < 8; ++i) {
    int blocks = (wn[i] + 255) / 256;
    hipLaunchKernelGGL(quant_kernel, dim3(blocks), dim3(256), 0, stream,
                       wts[i], wn[i], scales + i, wq16[i]);
  }
  // fold stalk maps into projection weights
  hipLaunchKernelGGL(compose_kernel, dim3(kD * kD / 256), dim3(256), 0, stream,
                     q_su, q_wq, eff_wq);
  hipLaunchKernelGGL(compose_kernel, dim3(kD * kD / 256), dim3(256), 0, stream,
                     q_sv, q_wk, eff_wk);

  // --- 2. LN1 ---
  hipLaunchKernelGGL(ln_kernel, dim3(kM), dim3(256), 0, stream,
                     x, n1g, n1b, xn1h, kD);

  // --- 3. Qs / Ks / V projections ---
  dim3 gD(kD / 128, kM / 128);
  hipLaunchKernelGGL(gemm_kernel, gD, dim3(256), 0, stream,
                     xn1h, eff_wq, (const float*)nullptr, (float*)nullptr, qs, kD, kD);
  hipLaunchKernelGGL(gemm_kernel, gD, dim3(256), 0, stream,
                     xn1h, eff_wk, (const float*)nullptr, (float*)nullptr, ks, kD, kD);
  hipLaunchKernelGGL(gemm_kernel, gD, dim3(256), 0, stream,
                     xn1h, q_wv, (const float*)nullptr, (float*)nullptr, vh, kD, kD);

  // --- 4. causal attention with 2-adic bias ---
  hipLaunchKernelGGL(attn_kernel, dim3(kT / 64, kH, kB), dim3(128), 0, stream,
                     qs, ks, vh, pscal, ao);

  // --- 5. output projection + residual -> x1 ---
  hipLaunchKernelGGL(gemm_kernel, gD, dim3(256), 0, stream,
                     ao, q_wo, x, x1, (_Float16*)nullptr, kD, kD);

  // --- 6. LN2 ---
  hipLaunchKernelGGL(ln_kernel, dim3(kM), dim3(256), 0, stream,
                     x1, n2g, n2b, xn2h, kD);

  // --- 7. MLP up ---
  dim3 gUp(kHid / 128, kM / 128);
  hipLaunchKernelGGL(gemm_kernel, gUp, dim3(256), 0, stream,
                     xn2h, q_wup, (const float*)nullptr, hbuf, (_Float16*)nullptr,
                     kHid, kD);

  // --- 8. rotate + LN + SiLU ---
  hipLaunchKernelGGL(rot_ln_silu_kernel, dim3(kM), dim3(256), 0, stream,
                     hbuf, theta, mng, mnb, acth);

  // --- 9. MLP down + residual -> out ---
  hipLaunchKernelGGL(gemm_kernel, gD, dim3(256), 0, stream,
                     acth, q_wdn, x1, out, (_Float16*)nullptr, kD, kHid);
}
